// ViTBlock_48687749267844
// MI455X (gfx1250) — matrix operand order repaired and checked by reading, not compiled, not run on hardware
//
#include <hip/hip_runtime.h>
#include <hip/hip_bf16.h>

typedef _Float16 f16;
typedef __attribute__((ext_vector_type(16))) _Float16     v16h;
typedef __attribute__((ext_vector_type(8)))  _Float16     v8h;
typedef __attribute__((ext_vector_type(8)))  float        v8f;
typedef __attribute__((ext_vector_type(4)))  float        v4f;
typedef __attribute__((ext_vector_type(4)))  unsigned int v4u;
typedef __attribute__((ext_vector_type(8)))  unsigned int v8u;

#define BDIM   16
#define CCH    384
#define NPIX   1024
#define NHD    8
#define HD     48
#define HDP    64
#define MIDC   384
#define HIDC   1536
#define PTOT   (BDIM * NPIX)   // 16384 pixels total

// ---------------------------------------------------------------------------
// WMMA helpers (CDNA5 16x16x32 f16 -> f32)
// ---------------------------------------------------------------------------
__device__ __forceinline__ v8f wmma_f16f32(v16h a, v16h b, v8f c) {
    return __builtin_amdgcn_wmma_f32_16x16x32_f16(false, a, false, b, (short)0, c,
                                                  false, false);
}

// A-matrix 16x32 f16 fragment. ISA layout: lanes 0-15 row M=lane, K pairs
// {0..7} in v0-3 and {16..23} in v4-7; lanes 16-31 same rows, K {8..15, 24..31}.
// base points at &A[row0][k0]; rows contiguous in K with stride ld.
__device__ __forceinline__ v16h frag_a16(const f16* base, int ld, int lane) {
    const int m  = lane & 15;
    const int kb = (lane < 16) ? 0 : 8;
    const f16* p = base + (size_t)m * ld + kb;
    v4u lo = *reinterpret_cast<const v4u*>(p);        // K = kb .. kb+7
    v4u hi = *reinterpret_cast<const v4u*>(p + 16);   // K = kb+16 .. kb+23
    v8u u  = __builtin_shufflevector(lo, hi, 0, 1, 2, 3, 4, 5, 6, 7);
    return __builtin_bit_cast(v16h, u);
}

// B-matrix 32x16 f16 fragment. Lanes 0-15 hold K {0..7, 16..23} of column N=lane,
// lanes 16-31 hold K {8..15, 24..31} (same K order as the A fragment).
// base points at &B[k0][col0]; each column's K run contiguous, stride ld
// between columns.
__device__ __forceinline__ v16h frag_b16(const f16* base, int ld, int lane) {
    const int n  = lane & 15;
    const int kb = (lane < 16) ? 0 : 8;
    const f16* p = base + (size_t)n * ld + kb;
    v4u lo = *reinterpret_cast<const v4u*>(p);        // K = kb .. kb+7
    v4u hi = *reinterpret_cast<const v4u*>(p + 16);   // K = kb+16 .. kb+23
    v8u u  = __builtin_shufflevector(lo, hi, 0, 1, 2, 3, 4, 5, 6, 7);
    return __builtin_bit_cast(v16h, u);
}

__device__ __forceinline__ v8f vzero8() {
    v8f z;
#pragma unroll
    for (int i = 0; i < 8; ++i) z[i] = 0.0f;
    return z;
}

// ---------------------------------------------------------------------------
// Elementwise / layout kernels
// ---------------------------------------------------------------------------
__global__ __launch_bounds__(256) void cvt_f16_kernel(const float* __restrict__ in,
                                                      f16* __restrict__ out, int n) {
    int i = blockIdx.x * 256 + threadIdx.x;
    if (i < n) out[i] = (f16)in[i];
}

// [batch][R][Ccol] -> [batch][Ccol][R], R and Ccol multiples of 32.
__global__ __launch_bounds__(256) void transpose_f32(const float* __restrict__ in,
                                                     float* __restrict__ out,
                                                     int R, int Ccol) {
    __shared__ float tile[32][33];
    const int b   = blockIdx.z;
    const float* inb = in + (size_t)b * R * Ccol;
    float* outb      = out + (size_t)b * R * Ccol;
    const int c0 = blockIdx.x * 32;
    const int r0 = blockIdx.y * 32;
    const int tx = threadIdx.x, ty = threadIdx.y;   // blockDim = (32, 8)
#pragma unroll
    for (int i = 0; i < 32; i += 8)
        tile[ty + i][tx] = inb[(size_t)(r0 + ty + i) * Ccol + c0 + tx];
    __syncthreads();
#pragma unroll
    for (int i = 0; i < 32; i += 8)
        outb[(size_t)(c0 + ty + i) * R + r0 + tx] = tile[tx][ty + i];
}

// LayerNorm over channel dim, pixel-major input [P, 384] f32 -> f16.
// One wave32 per pixel; 12 channels per lane; shuffle reductions.
__global__ __launch_bounds__(256) void ln_kernel(const float* __restrict__ in,
                                                 const float* __restrict__ g,
                                                 const float* __restrict__ bb,
                                                 f16* __restrict__ out) {
    const int lane = threadIdx.x & 31;
    const int pix  = blockIdx.x * 8 + (threadIdx.x >> 5);
    const float* px = in + (size_t)pix * CCH;
    float v[12];
    float s = 0.0f;
#pragma unroll
    for (int i = 0; i < 12; ++i) { v[i] = px[lane + 32 * i]; s += v[i]; }
#pragma unroll
    for (int off = 16; off >= 1; off >>= 1) s += __shfl_xor(s, off, 32);
    const float mean = s * (1.0f / CCH);
    float s2 = 0.0f;
#pragma unroll
    for (int i = 0; i < 12; ++i) { float d = v[i] - mean; s2 += d * d; }
#pragma unroll
    for (int off = 16; off >= 1; off >>= 1) s2 += __shfl_xor(s2, off, 32);
    const float rstd = rsqrtf(s2 * (1.0f / CCH) + 1e-5f);
    f16* op = out + (size_t)pix * CCH;
#pragma unroll
    for (int i = 0; i < 12; ++i) {
        int c = lane + 32 * i;
        op[c] = (f16)((v[i] - mean) * rstd * g[c] + bb[c]);
    }
}

// q_pm [P,384] f16, kv_pm [P,768] f16 ->
//   qh/kh [B,H,N,64] f16 (d padded 48->64 with zeros), vT [B,H,48,N] f16.
__global__ __launch_bounds__(256) void split_qkv_kernel(const f16* __restrict__ q_pm,
                                                        const f16* __restrict__ kv_pm,
                                                        f16* __restrict__ qh,
                                                        f16* __restrict__ kh,
                                                        f16* __restrict__ vT) {
    const int idx = blockIdx.x * 256 + threadIdx.x;   // 2^23 total
    const int d = idx & 63;
    const int n = (idx >> 6) & 1023;
    const int h = (idx >> 16) & 7;
    const int b = idx >> 19;
    const size_t pixrow = (size_t)(b * NPIX + n);
    f16 qv = (f16)0, kv = (f16)0;
    if (d < HD) {
        qv = q_pm[pixrow * MIDC + h * HD + d];
        kv = kv_pm[pixrow * (2 * MIDC) + h * HD + d];
        vT[((size_t)(b * NHD + h) * HD + d) * NPIX + n] =
            kv_pm[pixrow * (2 * MIDC) + MIDC + h * HD + d];
    }
    qh[idx] = qv;
    kh[idx] = kv;
}

// bias_full[h][n][m] = rpb_table[rel_index[n][m] * NH + h]
__global__ __launch_bounds__(256) void bias_expand_kernel(const float* __restrict__ rpb,
                                                          const int* __restrict__ rel,
                                                          float* __restrict__ biasf) {
    const int idx = blockIdx.x * 256 + threadIdx.x;   // 2^23 total
    const int m = idx & 1023;
    const int n = (idx >> 10) & 1023;
    const int h = idx >> 20;
    biasf[idx] = rpb[(size_t)rel[n * NPIX + m] * NHD + h];
}

// ---------------------------------------------------------------------------
// Generic pixel-major WMMA GEMM: out[p][o] = act(W[o][:] . X[p][:] + bias[o] (+resid))
// Wave tile 64(M) x 64(N); block = 4 waves in 2x2 -> 128x128 block tile.
// Per K-step: 16 x b128 loads for 16 WMMAs (1:1 vmem:wmma). K multiple of 32.
// EPI: 0 -> f16 out, 1 -> f32 out + residual, 2 -> exact GELU -> f16 out.
// ---------------------------------------------------------------------------
template <int EPI>
__global__ __launch_bounds__(128) void gemm_pm(const f16* __restrict__ W,
                                               const f16* __restrict__ X,
                                               const float* __restrict__ bias,
                                               const float* __restrict__ resid,
                                               void* __restrict__ outv,
                                               int O, int K) {
    const int lane = threadIdx.x & 31;
    const int wave = threadIdx.x >> 5;
    const int m0 = blockIdx.x * 128 + (wave & 1) * 64;
    const int n0 = blockIdx.y * 128 + (wave >> 1) * 64;

    v8f acc[4][4];
#pragma unroll
    for (int i = 0; i < 4; ++i)
#pragma unroll
        for (int j = 0; j < 4; ++j) acc[i][j] = vzero8();

    for (int k0 = 0; k0 < K; k0 += 32) {
        v16h a[4], b[4];
#pragma unroll
        for (int i = 0; i < 4; ++i)
            a[i] = frag_a16(W + (size_t)(m0 + 16 * i) * K + k0, K, lane);
#pragma unroll
        for (int j = 0; j < 4; ++j)
            b[j] = frag_b16(X + (size_t)(n0 + 16 * j) * K + k0, K, lane);
#pragma unroll
        for (int i = 0; i < 4; ++i)
#pragma unroll
            for (int j = 0; j < 4; ++j)
                acc[i][j] = wmma_f16f32(a[i], b[j], acc[i][j]);
    }

    const int hi = lane >> 4;
    const int nl = lane & 15;
#pragma unroll
    for (int i = 0; i < 4; ++i) {
#pragma unroll
        for (int j = 0; j < 4; ++j) {
            const int mb = m0 + 16 * i + 8 * hi;       // 8 consecutive o values
            const int n  = n0 + 16 * j + nl;
            if (EPI == 1) {
                float* op       = (float*)outv + (size_t)n * O + mb;
                const float* rp = resid + (size_t)n * O + mb;
                v4f lo, hv;
#pragma unroll
                for (int r = 0; r < 8; ++r) {
                    float t = acc[i][j][r] + bias[mb + r] + rp[r];
                    if (r < 4) lo[r] = t; else hv[r - 4] = t;
                }
                *reinterpret_cast<v4f*>(op)     = lo;
                *reinterpret_cast<v4f*>(op + 4) = hv;
            } else {
                v8h hvec;
#pragma unroll
                for (int r = 0; r < 8; ++r) {
                    float t = acc[i][j][r] + bias[mb + r];
                    if (EPI == 2) t = 0.5f * t * (1.0f + erff(t * 0.70710678118654752f));
                    hvec[r] = (f16)t;
                }
                *reinterpret_cast<v8h*>((f16*)outv + (size_t)n * O + mb) = hvec;
            }
        }
    }
}

// ---------------------------------------------------------------------------
// Flash attention: one wave per (b, head, 16-row Q tile). Keys in steps of 32.
// S = Q K^T (4 WMMAs), online softmax (row-max butterfly only), P restaged via
// LDS into A-layout, out += P V (3 WMMAs) and running row-sum accumulated by a
// 4th WMMA against a constant ones-column B fragment (no loads, no shuffles).
// Scores never leave the WGP.
// ---------------------------------------------------------------------------
__global__ __launch_bounds__(256) void attn_kernel(const f16* __restrict__ qh,
                                                   const f16* __restrict__ kh,
                                                   const f16* __restrict__ vT,
                                                   const float* __restrict__ biasf,
                                                   f16* __restrict__ outp) {
    __shared__ __align__(16) f16 plds[8][16 * 32];
    const int lane = threadIdx.x & 31;
    const int wave = threadIdx.x >> 5;
    const int gwid = blockIdx.x * 8 + wave;
    const int qt = gwid & 63;
    const int h  = (gwid >> 6) & 7;
    const int b  = gwid >> 9;
    const int n0 = qt * 16;

    const f16* qhh = qh + (size_t)(b * NHD + h) * NPIX * HDP;
    const f16* khh = kh + (size_t)(b * NHD + h) * NPIX * HDP;
    const f16* vTh = vT + (size_t)(b * NHD + h) * HD * NPIX;
    const float* bh = biasf + (size_t)h * NPIX * NPIX;
    f16* myp = &plds[wave][0];

    const int hi = lane >> 4;
    const int nl = lane & 15;

    // Constant B fragment: B[k][n] = (n == 0) for all k -> column 0 of the
    // product is the row-sum of A. Lane pattern only; no memory traffic.
    v16h onesb;
    {
        const f16 v1 = (nl == 0) ? (f16)1.0f : (f16)0.0f;
#pragma unroll
        for (int e = 0; e < 16; ++e) onesb[e] = v1;
    }

    const v16h qa0 = frag_a16(qhh + (size_t)n0 * HDP + 0, HDP, lane);
    const v16h qa1 = frag_a16(qhh + (size_t)n0 * HDP + 32, HDP, lane);

    v8f o0 = vzero8(), o1 = vzero8(), o2 = vzero8(), o3 = vzero8();
    float mrow[8];
#pragma unroll
    for (int r = 0; r < 8; ++r) mrow[r] = -1e30f;

    const float scale = 0.14433756729740643f;   // 1/sqrt(48)

    for (int m0 = 0; m0 < NPIX; m0 += 32) {
        // ---- scores: two 16x16 tiles, K = 64 (padded d) ----
        v8f s0 = vzero8(), s1 = vzero8();
        s0 = wmma_f16f32(qa0, frag_b16(khh + (size_t)m0 * HDP + 0, HDP, lane), s0);
        s0 = wmma_f16f32(qa1, frag_b16(khh + (size_t)m0 * HDP + 32, HDP, lane), s0);
        s1 = wmma_f16f32(qa0, frag_b16(khh + (size_t)(m0 + 16) * HDP + 0, HDP, lane), s1);
        s1 = wmma_f16f32(qa1, frag_b16(khh + (size_t)(m0 + 16) * HDP + 32, HDP, lane), s1);

        // ---- bias + online softmax (rows split across 16-lane halves) ----
#pragma unroll
        for (int r = 0; r < 8; ++r) {
            const int row = n0 + r + 8 * hi;
            const float x0 = s0[r] * scale + bh[(size_t)row * NPIX + m0 + nl];
            const float x1 = s1[r] * scale + bh[(size_t)row * NPIX + m0 + 16 + nl];
            float tmax = fmaxf(x0, x1);
#pragma unroll
            for (int off = 8; off >= 1; off >>= 1)
                tmax = fmaxf(tmax, __shfl_xor(tmax, off, 32));
            const float nm   = fmaxf(mrow[r], tmax);
            const float corr = __expf(mrow[r] - nm);
            mrow[r] = nm;
            const float p0 = __expf(x0 - nm);
            const float p1 = __expf(x1 - nm);
            o0[r] *= corr; o1[r] *= corr; o2[r] *= corr; o3[r] *= corr;
            const int lrow = r + 8 * hi;
            myp[lrow * 32 + nl]      = (f16)p0;
            myp[lrow * 32 + 16 + nl] = (f16)p1;
        }
        asm volatile("s_wait_dscnt 0" ::: "memory");

        // ---- P (16x32) as A fragment from LDS; PV + row-sum accumulate ----
        const v16h pa = frag_a16(myp, 32, lane);
        o0 = wmma_f16f32(pa, frag_b16(vTh + (size_t)0 * NPIX + m0, NPIX, lane), o0);
        o1 = wmma_f16f32(pa, frag_b16(vTh + (size_t)16 * NPIX + m0, NPIX, lane), o1);
        o2 = wmma_f16f32(pa, frag_b16(vTh + (size_t)32 * NPIX + m0, NPIX, lane), o2);
        o3 = wmma_f16f32(pa, onesb, o3);   // column 0 = running row-sum
    }

    // ---- normalize by row-sum (broadcast from column 0 of o3) and write ----
#pragma unroll
    for (int r = 0; r < 8; ++r) {
        const float sr  = __shfl(o3[r], lane & 16, 32);   // lane 0 / 16 of group
        const float inv = 1.0f / sr;
        const size_t rowg = (size_t)(b * NPIX + n0 + r + 8 * hi);
        f16* dst = outp + rowg * MIDC + h * HD;
        dst[0 * 16 + nl] = (f16)(o0[r] * inv);
        dst[1 * 16 + nl] = (f16)(o1[r] * inv);
        dst[2 * 16 + nl] = (f16)(o2[r] * inv);
    }
}

// ---------------------------------------------------------------------------
// Host-side orchestration
// ---------------------------------------------------------------------------
extern "C" void kernel_launch(void* const* d_in, const int* in_sizes, int n_in,
                              void* d_out, int out_size, void* d_ws, size_t ws_size,
                              hipStream_t stream) {
    const float* x       = (const float*)d_in[0];
    const float* rpb     = (const float*)d_in[1];
    const int*   rel     = (const int*)d_in[2];
    const float* ln1_g   = (const float*)d_in[3];
    const float* ln1_b   = (const float*)d_in[4];
    const float* ln2_g   = (const float*)d_in[5];
    const float* ln2_b   = (const float*)d_in[6];
    const float* wq      = (const float*)d_in[7];
    const float* bq      = (const float*)d_in[8];
    const float* wkv     = (const float*)d_in[9];
    const float* bkv     = (const float*)d_in[10];
    const float* wproj   = (const float*)d_in[11];
    const float* bproj   = (const float*)d_in[12];
    const float* w1      = (const float*)d_in[13];
    const float* b1      = (const float*)d_in[14];
    const float* w2      = (const float*)d_in[15];
    const float* b2      = (const float*)d_in[16];
    float* out = (float*)d_out;

    // workspace carve-out (256B aligned slices)
    size_t off = 0;
    auto carve = [&](size_t bytes) -> void* {
        void* p = (char*)d_ws + off;
        off += (bytes + 255) & ~(size_t)255;
        return p;
    };
    float* x_pm    = (float*)carve((size_t)PTOT * CCH * 4);       // 25.2 MB
    f16*   h1      = (f16*)carve((size_t)PTOT * CCH * 2);
    f16*   wq16    = (f16*)carve((size_t)MIDC * CCH * 2);
    f16*   wkv16   = (f16*)carve((size_t)2 * MIDC * CCH * 2);
    f16*   wproj16 = (f16*)carve((size_t)CCH * MIDC * 2);
    f16*   w116    = (f16*)carve((size_t)HIDC * CCH * 2);
    f16*   w216    = (f16*)carve((size_t)CCH * HIDC * 2);
    f16*   q_pm    = (f16*)carve((size_t)PTOT * MIDC * 2);
    f16*   kv_pm   = (f16*)carve((size_t)PTOT * 2 * MIDC * 2);
    f16*   qhB     = (f16*)carve((size_t)BDIM * NHD * NPIX * HDP * 2);
    f16*   khB     = (f16*)carve((size_t)BDIM * NHD * NPIX * HDP * 2);
    f16*   vTB     = (f16*)carve((size_t)BDIM * NHD * HD * NPIX * 2);
    float* biasf   = (float*)carve((size_t)NHD * NPIX * NPIX * 4);  // 33.6 MB
    f16*   attn_o  = (f16*)carve((size_t)PTOT * MIDC * 2);
    float* x2_pm   = (float*)carve((size_t)PTOT * CCH * 4);
    f16*   h2      = (f16*)carve((size_t)PTOT * CCH * 2);
    f16*   m1h     = (f16*)carve((size_t)PTOT * HIDC * 2);
    float* x3_pm   = (float*)carve((size_t)PTOT * CCH * 4);

    const dim3 tb32(32, 8, 1);

    // 0) weights f32 -> f16
    cvt_f16_kernel<<<(MIDC * CCH + 255) / 256, 256, 0, stream>>>(wq, wq16, MIDC * CCH);
    cvt_f16_kernel<<<(2 * MIDC * CCH + 255) / 256, 256, 0, stream>>>(wkv, wkv16, 2 * MIDC * CCH);
    cvt_f16_kernel<<<(CCH * MIDC + 255) / 256, 256, 0, stream>>>(wproj, wproj16, CCH * MIDC);
    cvt_f16_kernel<<<(HIDC * CCH + 255) / 256, 256, 0, stream>>>(w1, w116, HIDC * CCH);
    cvt_f16_kernel<<<(CCH * HIDC + 255) / 256, 256, 0, stream>>>(w2, w216, CCH * HIDC);

    // 1) x [B,C,N] -> x_pm [B,N,C]
    transpose_f32<<<dim3(NPIX / 32, CCH / 32, BDIM), tb32, 0, stream>>>(x, x_pm, CCH, NPIX);

    // 2) LN1 -> h1 f16
    ln_kernel<<<PTOT / 8, 256, 0, stream>>>(x_pm, ln1_g, ln1_b, h1);

    // 3) q and kv projections (WMMA)
    gemm_pm<0><<<dim3(MIDC / 128, PTOT / 128), 128, 0, stream>>>(wq16, h1, bq, nullptr, q_pm, MIDC, CCH);
    gemm_pm<0><<<dim3(2 * MIDC / 128, PTOT / 128), 128, 0, stream>>>(wkv16, h1, bkv, nullptr, kv_pm, 2 * MIDC, CCH);

    // 4) head split / pad / V transpose
    split_qkv_kernel<<<(BDIM * NHD * NPIX * HDP) / 256, 256, 0, stream>>>(q_pm, kv_pm, qhB, khB, vTB);

    // 5) relative-position bias expansion [H,N,N]
    bias_expand_kernel<<<(NHD * NPIX * NPIX) / 256, 256, 0, stream>>>(rpb, rel, biasf);

    // 6) flash attention (WMMA QK^T + softmax + WMMA PV + WMMA row-sums)
    attn_kernel<<<(BDIM * NHD * (NPIX / 16)) / 8, 256, 0, stream>>>(qhB, khB, vTB, biasf, attn_o);

    // 7) output projection + residual -> x2_pm (f32)
    gemm_pm<1><<<dim3(CCH / 128, PTOT / 128), 128, 0, stream>>>(wproj16, attn_o, bproj, x_pm, x2_pm, CCH, MIDC);

    // 8) LN2 -> h2 f16
    ln_kernel<<<PTOT / 8, 256, 0, stream>>>(x2_pm, ln2_g, ln2_b, h2);

    // 9) MLP: fc1 + exact GELU -> f16, fc2 + residual -> x3_pm
    gemm_pm<2><<<dim3(HIDC / 128, PTOT / 128), 128, 0, stream>>>(w116, h2, b1, nullptr, m1h, HIDC, CCH);
    gemm_pm<1><<<dim3(CCH / 128, PTOT / 128), 128, 0, stream>>>(w216, m1h, b2, x2_pm, x3_pm, CCH, HIDC);

    // 10) x3_pm [B,N,C] -> out [B,C,H,W]
    transpose_f32<<<dim3(CCH / 32, NPIX / 32, BDIM), tb32, 0, stream>>>(x3_pm, out, NPIX, CCH);
}
